// CvtSelfAttention_9801115369834
// MI455X (gfx1250) — matrix operand order repaired and checked by reading, not compiled, not run on hardware
//
#include <hip/hip_runtime.h>

typedef __attribute__((ext_vector_type(16))) _Float16 v16h;
typedef __attribute__((ext_vector_type(8)))  float    v8f;

#define BATCH 64
#define CH    384
#define NHEAD 6
#define HD    64
#define LQ    1025
#define LQP   1040   // 65 * 16
#define LKP   288    // 9 * 32 (real tokens: 257)
#define LKREAL 257
#define EPSBN 1e-5f

// ---------------- WMMA fragment helpers (wave32, ISA 7.12.2 layouts) ----------

// A-matrix 16x32 f16: lane L -> M = L&15 ; halves per VGPR hold K pairs.
// VGPR v: K0 = (v>>2)*16 + ((L>>4)*8) + (v&3)*2
__device__ inline v16h load_a_frag(const _Float16* __restrict__ src, int ld) {
    int lane = threadIdx.x & 31;
    int m  = lane & 15;
    int kh = (lane >> 4) * 8;
    v16h a;
#pragma unroll
    for (int v = 0; v < 8; ++v) {
        int k0 = (v >> 2) * 16 + kh + (v & 3) * 2;
        a[2 * v]     = src[m * ld + k0];
        a[2 * v + 1] = src[m * ld + k0 + 1];
    }
    return a;
}

// B-matrix 32x16 f16: lane L -> N = L&15 ; VGPR v holds K = (v>>2)*16 + kset + (v&3)*2, +1
// element B[k][n] at src[k*strideK + n*strideN]
__device__ inline v16h load_b_frag(const _Float16* __restrict__ src,
                                   int strideK, int strideN) {
    int lane = threadIdx.x & 31;
    int n    = lane & 15;
    int kset = (lane >> 4) * 8;
    v16h bf;
#pragma unroll
    for (int v = 0; v < 8; ++v) {
        int k = (v >> 2) * 16 + kset + (v & 3) * 2;
        bf[2 * v]     = src[k * strideK + n * strideN];
        bf[2 * v + 1] = src[(k + 1) * strideK + n * strideN];
    }
    return bf;
}

__device__ inline v8f wmma16(v16h a, v16h b, v8f c) {
    return __builtin_amdgcn_wmma_f32_16x16x32_f16(
        /*neg_a=*/false, a, /*neg_b=*/false, b,
        /*c_mod=*/(short)0, c, /*reuse_a=*/false, /*reuse_b=*/false);
}

// ---------------- Stage 1: depthwise conv 3x3 + BN -> fp16 tokens -------------

__global__ void dwconv_bn(const float* __restrict__ hs, const float* __restrict__ w,
                          const float* __restrict__ gamma, const float* __restrict__ beta,
                          const float* __restrict__ mean,  const float* __restrict__ var,
                          _Float16* __restrict__ dst, int stride, int Wo, int Lpad) {
    int i = blockIdx.x * 256 + threadIdx.x;
    int total = BATCH * CH * Wo * Wo;
    if (i >= total) return;
    int c = i % CH;  int t = i / CH;
    int ox = t % Wo; t /= Wo;
    int oy = t % Wo; int b = t / Wo;
    float inv = gamma[c] * rsqrtf(var[c] + EPSBN);
    float sh  = beta[c] - mean[c] * inv;
    float acc = 0.f;
#pragma unroll
    for (int ky = 0; ky < 3; ++ky) {
        int iy = oy * stride - 1 + ky;
        if (iy < 0 || iy >= 32) continue;
#pragma unroll
        for (int kx = 0; kx < 3; ++kx) {
            int ix = ox * stride - 1 + kx;
            if (ix < 0 || ix >= 32) continue;
            acc += hs[((size_t)b * LQ + 1 + iy * 32 + ix) * CH + c] * w[c * 9 + ky * 3 + kx];
        }
    }
    dst[((size_t)b * Lpad + 1 + oy * Wo + ox) * CH + c] = (_Float16)(acc * inv + sh);
}

__global__ void pack_cls(const float* __restrict__ hs, _Float16* __restrict__ Xq,
                         _Float16* __restrict__ Xk, _Float16* __restrict__ Xv) {
    int i = blockIdx.x * 256 + threadIdx.x;
    if (i >= BATCH * CH) return;
    int c = i % CH, b = i / CH;
    _Float16 v = (_Float16)hs[(size_t)b * LQ * CH + c];
    Xq[(size_t)b * LQP * CH + c] = v;
    Xk[(size_t)b * LKP * CH + c] = v;
    Xv[(size_t)b * LKP * CH + c] = v;
}

__global__ void f32_to_f16(const float* __restrict__ src, _Float16* __restrict__ dst, int n) {
    int i = blockIdx.x * 256 + threadIdx.x;
    if (i < n) dst[i] = (_Float16)src[i];
}

// ---------------- Stage 2: projection GEMM (WMMA) -----------------------------
// Out[b, l, o] = sum_c X[b,l,c] * W[o,c] + bias[o]
// vTranspose == 0 : store per-head (b,h,l,d) f16   (Q, K)
// vTranspose == 1 : store per-head (b,h,d,l) f16   (V, for contiguous ctx B-frags)

__global__ __launch_bounds__(32) void proj_kernel(const _Float16* __restrict__ X,
                                                  const _Float16* __restrict__ W,
                                                  const float* __restrict__ bias,
                                                  _Float16* __restrict__ Out, int Lpad,
                                                  int vTranspose) {
    int mtiles = Lpad / 16;
    int bid = blockIdx.x;
    int nt = bid % 24; bid /= 24;         // 384 / 16 output tiles
    int mt = bid % mtiles; int b = bid / mtiles;
    const _Float16* Xb = X + ((size_t)b * Lpad + mt * 16) * CH;
    v8f acc = {};
#pragma unroll 1
    for (int ks = 0; ks < 12; ++ks) {     // K = 384 = 12 * 32
        v16h a  = load_a_frag(Xb + ks * 32, CH);
        // B[k][n] = W[(nt*16+n)*384 + ks*32 + k]
        v16h bf = load_b_frag(W + (size_t)nt * 16 * CH + ks * 32, 1, CH);
        acc = wmma16(a, bf, acc);
    }
    int lane  = threadIdx.x & 31;
    int n     = lane & 15;
    int mbase = (lane >> 4) * 8;
    float bv  = bias[nt * 16 + n];
    int h  = nt / 4;                      // 64 = 4 tiles per head
    int d0 = (nt % 4) * 16;
    if (!vTranspose) {
        _Float16* Ob = Out + ((size_t)(b * NHEAD + h) * Lpad + mt * 16 + mbase) * HD + d0 + n;
#pragma unroll
        for (int r = 0; r < 8; ++r)
            Ob[(size_t)r * HD] = (_Float16)(acc[r] + bv);
    } else {
        // (b,h,d,l): 8 consecutive halves along l per lane -> 16B store
        _Float16* Ob = Out + ((size_t)(b * NHEAD + h) * HD + d0 + n) * Lpad + mt * 16 + mbase;
#pragma unroll
        for (int r = 0; r < 8; ++r)
            Ob[r] = (_Float16)(acc[r] + bv);
    }
}

// ---------------- Stage 3: fused attention (WMMA + LDS softmax) ---------------
// One wave per (b, h, 16 q-rows). S(16x288) -> softmax(t<257) -> P f16 -> PV.

__global__ __launch_bounds__(32) void attn_kernel(const _Float16* __restrict__ Q,
                                                  const _Float16* __restrict__ K,
                                                  const _Float16* __restrict__ Vt,
                                                  float* __restrict__ Out) {
    __shared__ float    Sl[16 * LKP];
    __shared__ _Float16 Pl[16 * LKP];
    int bid = blockIdx.x;
    int mt = bid % (LQP / 16); bid /= (LQP / 16);
    int h  = bid % NHEAD;      int b = bid / NHEAD;
    int l0 = mt * 16;
    const _Float16* Qb  = Q  + ((size_t)(b * NHEAD + h) * LQP + l0) * HD;
    const _Float16* Kb  = K  + (size_t)(b * NHEAD + h) * LKP * HD;
    const _Float16* VbT = Vt + (size_t)(b * NHEAD + h) * HD * LKP;   // (d, t) layout
    int lane  = threadIdx.x & 31;
    int n     = lane & 15;
    int mbase = (lane >> 4) * 8;

    // Q tile A-fragments, K-dim = head_dim (2 steps of 32)
    v16h aq0 = load_a_frag(Qb, HD);
    v16h aq1 = load_a_frag(Qb + 32, HD);

    // S = Q * K^T   (B[k=d][n=t] = K[t*64 + d] -> strideK=1, strideN=64)
#pragma unroll 1
    for (int nt = 0; nt < LKP / 16; ++nt) {
        if (nt + 1 < LKP / 16)
            __builtin_prefetch(Kb + (size_t)(nt + 1) * 16 * HD, 0, 3);
        v8f acc = {};
        v16h b0 = load_b_frag(Kb + (size_t)nt * 16 * HD, 1, HD);
        acc = wmma16(aq0, b0, acc);
        v16h b1 = load_b_frag(Kb + (size_t)nt * 16 * HD + 32, 1, HD);
        acc = wmma16(aq1, b1, acc);
#pragma unroll
        for (int r = 0; r < 8; ++r)
            Sl[(mbase + r) * LKP + nt * 16 + n] = acc[r];
    }
    __syncthreads();

    // softmax over real tokens (t < 257), scale = C^-0.5 per reference.
    // Full wave: lane l and lane l+16 share row (l&15), split the column range,
    // combine running max / sum via shfl_xor(16).
    {
        const float scale = 0.05103103630798288f;  // 1/sqrt(384)
        int row  = lane & 15;
        int half = lane >> 4;
        int t0 = half ? 128 : 0;
        int t1 = half ? LKREAL : 128;
        float*    rowp = Sl + row * LKP;
        _Float16* prow = Pl + row * LKP;
        float mx = -3.4e38f;
        for (int t = t0; t < t1; ++t) mx = fmaxf(mx, rowp[t] * scale);
        mx = fmaxf(mx, __shfl_xor(mx, 16, 32));
        float sum = 0.f;
        for (int t = t0; t < t1; ++t) {
            float e = __expf(rowp[t] * scale - mx);
            rowp[t] = e; sum += e;
        }
        sum += __shfl_xor(sum, 16, 32);
        float inv = 1.f / sum;
        for (int t = t0; t < t1; ++t) prow[t] = (_Float16)(rowp[t] * inv);
        if (half)
            for (int t = LKREAL; t < LKP; ++t) prow[t] = (_Float16)0.f;
    }
    __syncthreads();

    // ctx = P * V   (A[m][k=t] from LDS; B[k=t][n=d] = Vt[d*LKP + t]
    //                -> strideK=1, strideN=LKP : contiguous per-lane b128 loads)
    v8f accd[4] = {};
#pragma unroll 1
    for (int tb = 0; tb < LKP / 32; ++tb) {
        if (tb + 1 < LKP / 32)
            __builtin_prefetch(VbT + (size_t)(tb + 1) * 32, 0, 3);
        v16h ap = load_a_frag(Pl + tb * 32, LKP);
#pragma unroll
        for (int dt = 0; dt < 4; ++dt) {
            v16h bv = load_b_frag(VbT + (size_t)dt * 16 * LKP + tb * 32, 1, LKP);
            accd[dt] = wmma16(ap, bv, accd[dt]);
        }
    }
    // write (b, l, h*64+d) f32, drop padded rows
#pragma unroll
    for (int dt = 0; dt < 4; ++dt)
#pragma unroll
        for (int r = 0; r < 8; ++r) {
            int gl = l0 + mbase + r;
            if (gl < LQ)
                Out[((size_t)b * LQ + gl) * CH + h * HD + dt * 16 + n] = accd[dt][r];
        }
}

// ---------------- Host launch --------------------------------------------------

extern "C" void kernel_launch(void* const* d_in, const int* in_sizes, int n_in,
                              void* d_out, int out_size, void* d_ws, size_t ws_size,
                              hipStream_t stream) {
    const float* hs = (const float*)d_in[0];
    // per-projection inputs: [1..7]=q, [8..14]=k, [15..21]=v
    const float* convw[3] = {(const float*)d_in[1],  (const float*)d_in[8],  (const float*)d_in[15]};
    const float* gamma[3] = {(const float*)d_in[2],  (const float*)d_in[9],  (const float*)d_in[16]};
    const float* beta [3] = {(const float*)d_in[3],  (const float*)d_in[10], (const float*)d_in[17]};
    const float* mean [3] = {(const float*)d_in[4],  (const float*)d_in[11], (const float*)d_in[18]};
    const float* var  [3] = {(const float*)d_in[5],  (const float*)d_in[12], (const float*)d_in[19]};
    const float* Wmat [3] = {(const float*)d_in[6],  (const float*)d_in[13], (const float*)d_in[20]};
    const float* bias [3] = {(const float*)d_in[7],  (const float*)d_in[14], (const float*)d_in[21]};

    // workspace layout (fp16)
    const size_t szXq = (size_t)BATCH * LQP * CH;          // 25.56M elems
    const size_t szXk = (size_t)BATCH * LKP * CH;          //  7.08M elems
    const size_t szW  = (size_t)CH * CH;
    _Float16* Xq  = (_Float16*)d_ws;
    _Float16* Xk  = Xq + szXq;
    _Float16* Xv  = Xk + szXk;
    _Float16* W16 = Xv + szXk;                             // 3 weight matrices
    _Float16* Q16 = W16 + 3 * szW;                         // (b,h,LQP,64)
    _Float16* K16 = Q16 + szXq;                            // (b,h,LKP,64)
    _Float16* V16 = K16 + szXk;                            // (b,h,64,LKP) transposed

    // zero token buffers (covers cls row before overwrite + padded rows)
    hipMemsetAsync(Xq, 0, (szXq + 2 * szXk) * sizeof(_Float16), stream);

    // convert projection weights to fp16
    for (int p = 0; p < 3; ++p)
        f32_to_f16<<<((int)szW + 255) / 256, 256, 0, stream>>>(Wmat[p], W16 + p * szW, (int)szW);

    // cls token into row 0 of each token buffer
    pack_cls<<<(BATCH * CH + 255) / 256, 256, 0, stream>>>(hs, Xq, Xk, Xv);

    // depthwise conv + BN
    {
        int tq = BATCH * CH * 32 * 32;
        dwconv_bn<<<(tq + 255) / 256, 256, 0, stream>>>(hs, convw[0], gamma[0], beta[0],
                                                        mean[0], var[0], Xq, 1, 32, LQP);
        int tk = BATCH * CH * 16 * 16;
        dwconv_bn<<<(tk + 255) / 256, 256, 0, stream>>>(hs, convw[1], gamma[1], beta[1],
                                                        mean[1], var[1], Xk, 2, 16, LKP);
        dwconv_bn<<<(tk + 255) / 256, 256, 0, stream>>>(hs, convw[2], gamma[2], beta[2],
                                                        mean[2], var[2], Xv, 2, 16, LKP);
    }

    // WMMA projections -> per-head fp16 Q/K (row-major), V (transposed)
    proj_kernel<<<BATCH * (LQP / 16) * 24, 32, 0, stream>>>(Xq, W16 + 0 * szW, bias[0], Q16, LQP, 0);
    proj_kernel<<<BATCH * (LKP / 16) * 24, 32, 0, stream>>>(Xk, W16 + 1 * szW, bias[1], K16, LKP, 0);
    proj_kernel<<<BATCH * (LKP / 16) * 24, 32, 0, stream>>>(Xv, W16 + 2 * szW, bias[2], V16, LKP, 1);

    // fused attention
    attn_kernel<<<BATCH * NHEAD * (LQP / 16), 32, 0, stream>>>(Q16, K16, V16, (float*)d_out);
}
